// Net_8787503087948
// MI455X (gfx1250) — matrix operand order repaired and checked by reading, not compiled, not run on hardware
//
#include <hip/hip_runtime.h>
#include <hip/hip_bf16.h>

// ---------------- CDNA5 WMMA types ----------------
typedef __attribute__((ext_vector_type(16))) __bf16 v16bf;
typedef __attribute__((ext_vector_type(8)))  float  v8f;

struct __align__(16) uq { unsigned int a, b, c, d; };
union Frag { uq q[2]; v16bf v; };

#define N_NODES 50000
#define N_EDGES 400000
#define NGRAPH  64

__device__ __forceinline__ unsigned short f2bf(float f) {
  unsigned int u = __float_as_uint(f);
  u += 0x7FFFu + ((u >> 16) & 1u);       // round-to-nearest-even
  return (unsigned short)(u >> 16);
}
__device__ __forceinline__ float elu1(float v) { return v > 0.f ? v : __expf(v) - 1.f; }

// ---------------- zero scratch ----------------
__global__ void zero_kernel(float* __restrict__ p, int n) {
  for (int i = blockIdx.x * blockDim.x + threadIdx.x; i < n; i += gridDim.x * blockDim.x)
    p[i] = 0.f;
}

// ---------------- pack constant B matrices into wave32 WMMA B-fragment order ----------------
// B1[(k,i), o] = nn1_w2[k, i*16+o]; rows 320..351 = nn1_b2[i*16+o]  (K=352, N=16)
__global__ void prep_b1_kernel(const float* __restrict__ w2, const float* __restrict__ b2,
                               unsigned short* __restrict__ B1p) {
  int idx = blockIdx.x * blockDim.x + threadIdx.x;
  if (idx >= 11 * 32 * 16) return;
  int j = idx & 15, Lr = (idx >> 4) & 31, t = idx >> 9;
  int K = t * 32 + ((Lr >> 4) << 4) + j;   // lanes 0-15: K=tile..tile+15, lanes 16-31: +16
  int o = Lr & 15;
  float v;
  if (K < 320) { int k = K >> 5, i = K & 31; v = w2[k * 512 + i * 16 + o]; }
  else         { int i = K - 320;            v = b2[i * 16 + o]; }
  B1p[idx] = f2bf(v);
}

// B2[(k,i), o] = nn2_w2[k, i*32+o]; rows 160..175 = nn2_b2; 176..191 = 0 (K=192, N=32 -> 2 halves)
__global__ void prep_b2_kernel(const float* __restrict__ w2, const float* __restrict__ b2,
                               unsigned short* __restrict__ B2p) {
  int idx = blockIdx.x * blockDim.x + threadIdx.x;
  if (idx >= 6 * 2 * 32 * 16) return;
  int j = idx & 15, Lr = (idx >> 4) & 31, nh = (idx >> 9) & 1, t = idx >> 10;
  int K = t * 32 + ((Lr >> 4) << 4) + j;
  int o = nh * 16 + (Lr & 15);
  float v;
  if (K < 160)      { int k = K >> 4, i = K & 15; v = w2[k * 512 + i * 32 + o]; }
  else if (K < 176) { int i = K - 160;            v = b2[i * 32 + o]; }
  else              v = 0.f;
  B2p[idx] = f2bf(v);
}

// ---------------- layer-1 edge GEMM: Z[16,352] x B1[352,16] per edge-tile ----------------
__global__ __launch_bounds__(32) void edge1_kernel(
    const float* __restrict__ x, const int* __restrict__ ei, const float* __restrict__ ea,
    const float* __restrict__ w1, const float* __restrict__ b1,
    const unsigned short* __restrict__ B1p,
    float* __restrict__ agg1, float* __restrict__ cnt) {
  __shared__ unsigned short Z[16 * 352];   // bf16 Z tile, row = edge-in-tile
  __shared__ int dsts[16];
  const int L = threadIdx.x;
  const int m = L & 15, half = L >> 4;

  Frag B[11];                               // constant B resident in VGPRs
  #pragma unroll
  for (int t = 0; t < 11; ++t) {
    const uq* p = (const uq*)B1p + (t * 32 + L) * 2;
    B[t].q[0] = p[0]; B[t].q[1] = p[1];
  }
  float w1r[10], b1r[10];
  #pragma unroll
  for (int k = 0; k < 10; ++k) { w1r[k] = w1[k]; b1r[k] = b1[k]; }

  for (int tile = blockIdx.x; tile < N_EDGES / 16; tile += gridDim.x) {
    const int e = tile * 16 + m;
    const float a = ea[e];
    const int src = ei[e];
    if (half == 0) dsts[m] = ei[N_EDGES + e];

    float h[10];
    #pragma unroll
    for (int k = 0; k < 10; ++k) { float v = fmaf(a, w1r[k], b1r[k]); h[k] = v > 0.f ? v : 0.f; }

    float xi[16];                           // this lane's half of x[src]
    const float4* xp = (const float4*)(x + (size_t)src * 32 + half * 16);
    #pragma unroll
    for (int q = 0; q < 4; ++q) { float4 v = xp[q];
      xi[q*4+0]=v.x; xi[q*4+1]=v.y; xi[q*4+2]=v.z; xi[q*4+3]=v.w; }

    unsigned int* Zrow = (unsigned int*)&Z[m * 352];
    #pragma unroll
    for (int k = 0; k < 10; ++k) {
      #pragma unroll
      for (int i = 0; i < 16; i += 2) {
        unsigned int pk = (unsigned int)f2bf(h[k] * xi[i]) |
                          ((unsigned int)f2bf(h[k] * xi[i + 1]) << 16);
        Zrow[(k * 32 + half * 16 + i) >> 1] = pk;
      }
    }
    #pragma unroll
    for (int i = 0; i < 16; i += 2) {       // bias rows: raw x
      unsigned int pk = (unsigned int)f2bf(xi[i]) | ((unsigned int)f2bf(xi[i + 1]) << 16);
      Zrow[(320 + half * 16 + i) >> 1] = pk;
    }
    asm volatile("s_wait_dscnt 0" ::: "memory");  // cross-lane LDS handoff

    v8f acc = {};
    #pragma unroll
    for (int t = 0; t < 11; ++t) {
      Frag af;                              // A-fragment, contiguous k = 16*half + s (matches B1p)
      af.q[0] = *(const uq*)&Z[m * 352 + t * 32 + 16 * half];
      af.q[1] = *(const uq*)&Z[m * 352 + t * 32 + 16 * half + 8];
      acc = __builtin_amdgcn_wmma_f32_16x16x32_bf16(false, af.v, false, B[t].v,
                                                    (short)0, acc, false, false);
    }

    #pragma unroll
    for (int r = 0; r < 8; ++r)             // C layout: row = r + 8*half, col = L&15
      atomicAdd(&agg1[(size_t)dsts[r + 8 * half] * 16 + m], acc[r]);
    if (half == 0) atomicAdd(&cnt[dsts[m]], 1.0f);
  }
}

// ---------------- layer-2 edge GEMM: Z[16,192] x B2[192,32] ----------------
__global__ __launch_bounds__(32) void edge2_kernel(
    const float* __restrict__ h1, const int* __restrict__ ei, const float* __restrict__ ea,
    const float* __restrict__ w1, const float* __restrict__ b1,
    const unsigned short* __restrict__ B2p,
    float* __restrict__ agg2) {
  __shared__ unsigned short Z[16 * 192];
  __shared__ int dsts[16];
  const int L = threadIdx.x;
  const int m = L & 15, half = L >> 4;

  Frag B[6][2];
  #pragma unroll
  for (int t = 0; t < 6; ++t)
    #pragma unroll
    for (int nh = 0; nh < 2; ++nh) {
      const uq* p = (const uq*)B2p + ((t * 2 + nh) * 32 + L) * 2;
      B[t][nh].q[0] = p[0]; B[t][nh].q[1] = p[1];
    }
  float w1r[10], b1r[10];
  #pragma unroll
  for (int k = 0; k < 10; ++k) { w1r[k] = w1[k]; b1r[k] = b1[k]; }

  for (int tile = blockIdx.x; tile < N_EDGES / 16; tile += gridDim.x) {
    const int e = tile * 16 + m;
    const float a = ea[e];
    const int src = ei[e];
    if (half == 0) dsts[m] = ei[N_EDGES + e];

    float h[10];
    #pragma unroll
    for (int k = 0; k < 10; ++k) { float v = fmaf(a, w1r[k], b1r[k]); h[k] = v > 0.f ? v : 0.f; }

    float hs[16];
    const float4* hp = (const float4*)(h1 + (size_t)src * 16);
    #pragma unroll
    for (int q = 0; q < 4; ++q) { float4 v = hp[q];
      hs[q*4+0]=v.x; hs[q*4+1]=v.y; hs[q*4+2]=v.z; hs[q*4+3]=v.w; }

    unsigned int* Zrow = (unsigned int*)&Z[m * 192];
    if (half == 0) {                        // rows 0..95 : k=0..5
      #pragma unroll
      for (int p0 = 0; p0 < 96; p0 += 2) {
        unsigned int pk = (unsigned int)f2bf(h[p0 >> 4] * hs[p0 & 15]) |
                          ((unsigned int)f2bf(h[(p0 + 1) >> 4] * hs[(p0 + 1) & 15]) << 16);
        Zrow[p0 >> 1] = pk;
      }
    } else {                                // rows 96..159: k=6..9; 160..175: bias; 176..191: 0
      #pragma unroll
      for (int p0 = 0; p0 < 96; p0 += 2) {
        const int p = 96 + p0;
        float v0, v1;
        if (p0 < 64)      { v0 = h[p >> 4] * hs[p & 15]; v1 = h[(p + 1) >> 4] * hs[(p + 1) & 15]; }
        else if (p0 < 80) { v0 = hs[p0 - 64]; v1 = hs[p0 - 63]; }
        else              { v0 = 0.f; v1 = 0.f; }
        unsigned int pk = (unsigned int)f2bf(v0) | ((unsigned int)f2bf(v1) << 16);
        Zrow[p >> 1] = pk;
      }
    }
    asm volatile("s_wait_dscnt 0" ::: "memory");

    v8f acc0 = {}, acc1 = {};
    #pragma unroll
    for (int t = 0; t < 6; ++t) {
      Frag af;
      af.q[0] = *(const uq*)&Z[m * 192 + t * 32 + 16 * half];
      af.q[1] = *(const uq*)&Z[m * 192 + t * 32 + 16 * half + 8];
      acc0 = __builtin_amdgcn_wmma_f32_16x16x32_bf16(false, af.v, false, B[t][0].v,
                                                     (short)0, acc0, false, false);
      acc1 = __builtin_amdgcn_wmma_f32_16x16x32_bf16(false, af.v, false, B[t][1].v,
                                                     (short)0, acc1, false, false);
    }

    #pragma unroll
    for (int r = 0; r < 8; ++r) {
      const int d = dsts[r + 8 * half];
      atomicAdd(&agg2[(size_t)d * 32 + m],      acc0[r]);
      atomicAdd(&agg2[(size_t)d * 32 + 16 + m], acc1[r]);
    }
  }
}

// ---------------- node update 1: mean + x@root + bias, ELU ----------------
__global__ void node1_kernel(const float* __restrict__ x, const float* __restrict__ agg1,
                             const float* __restrict__ cnt, const float* __restrict__ root,
                             const float* __restrict__ bias, float* __restrict__ h1) {
  int idx = blockIdx.x * blockDim.x + threadIdx.x;
  if (idx >= N_NODES * 16) return;
  int n = idx >> 4, c = idx & 15;
  float mean = agg1[idx] / fmaxf(cnt[n], 1.0f);
  float dot = 0.f;
  #pragma unroll
  for (int i = 0; i < 32; ++i) dot = fmaf(x[(size_t)n * 32 + i], root[i * 16 + c], dot);
  h1[idx] = elu1(mean + dot + bias[c]);
}

// ---------------- node update 2 + global mean pool accumulation ----------------
__global__ void node2_pool_kernel(const float* __restrict__ h1, const float* __restrict__ agg2,
                                  const float* __restrict__ cnt, const float* __restrict__ root,
                                  const float* __restrict__ bias, const int* __restrict__ batch,
                                  float* __restrict__ gsum, float* __restrict__ gcnt) {
  int idx = blockIdx.x * blockDim.x + threadIdx.x;
  if (idx >= N_NODES * 32) return;
  int n = idx >> 5, c = idx & 31;
  float mean = agg2[idx] / fmaxf(cnt[n], 1.0f);
  float dot = 0.f;
  #pragma unroll
  for (int i = 0; i < 16; ++i) dot = fmaf(h1[(size_t)n * 16 + i], root[i * 32 + c], dot);
  float v = elu1(mean + dot + bias[c]);
  int g = batch[n];
  atomicAdd(&gsum[g * 32 + c], v);
  if (c == 0) atomicAdd(&gcnt[g], 1.0f);
}

// ---------------- head: fc1 -> ELU -> fc2 -> log_softmax ----------------
__global__ __launch_bounds__(64) void head_kernel(
    const float* __restrict__ gsum, const float* __restrict__ gcnt,
    const float* __restrict__ fc1w, const float* __restrict__ fc1b,
    const float* __restrict__ fc2w, const float* __restrict__ fc2b,
    float* __restrict__ out) {
  int gi = threadIdx.x;
  if (gi >= NGRAPH) return;
  float g[32];
  float ic = 1.f / fmaxf(gcnt[gi], 1.f);
  #pragma unroll
  for (int i = 0; i < 32; ++i) g[i] = gsum[gi * 32 + i] * ic;
  float l0 = fc2b[0], l1 = fc2b[1];
  for (int j = 0; j < 64; ++j) {
    float hj = fc1b[j];
    #pragma unroll
    for (int i = 0; i < 32; ++i) hj = fmaf(g[i], fc1w[i * 64 + j], hj);
    hj = elu1(hj);
    l0 = fmaf(hj, fc2w[j * 2 + 0], l0);
    l1 = fmaf(hj, fc2w[j * 2 + 1], l1);
  }
  float mx = fmaxf(l0, l1);
  float lse = mx + __logf(__expf(l0 - mx) + __expf(l1 - mx));
  out[gi * 2 + 0] = l0 - lse;
  out[gi * 2 + 1] = l1 - lse;
}

// ---------------- launch ----------------
extern "C" void kernel_launch(void* const* d_in, const int* in_sizes, int n_in,
                              void* d_out, int out_size, void* d_ws, size_t ws_size,
                              hipStream_t stream) {
  const float* x     = (const float*)d_in[0];
  const int*   ei    = (const int*)  d_in[1];
  const float* ea    = (const float*)d_in[2];
  const int*   batch = (const int*)  d_in[3];
  const float* n1w1  = (const float*)d_in[4];
  const float* n1b1  = (const float*)d_in[5];
  const float* n1w2  = (const float*)d_in[6];
  const float* n1b2  = (const float*)d_in[7];
  const float* root1 = (const float*)d_in[8];
  const float* bias1 = (const float*)d_in[9];
  const float* n2w1  = (const float*)d_in[10];
  const float* n2b1  = (const float*)d_in[11];
  const float* n2w2  = (const float*)d_in[12];
  const float* n2b2  = (const float*)d_in[13];
  const float* root2 = (const float*)d_in[14];
  const float* bias2 = (const float*)d_in[15];
  const float* fc1w  = (const float*)d_in[16];
  const float* fc1b  = (const float*)d_in[17];
  const float* fc2w  = (const float*)d_in[18];
  const float* fc2b  = (const float*)d_in[19];
  float* out = (float*)d_out;

  // workspace layout (floats)
  float* W    = (float*)d_ws;
  float* agg1 = W;                    // 800000
  float* agg2 = W + 800000;           // 1600000
  float* cnt  = W + 2400000;          // 50000
  float* gsum = W + 2450000;          // 2048
  float* gcnt = W + 2452048;          // 64
  float* h1   = W + 2452112;          // 800000
  unsigned short* B1p = (unsigned short*)(W + 3252112);  // 5632 bf16
  unsigned short* B2p = B1p + 5632;                      // 6144 bf16

  zero_kernel<<<2048, 256, 0, stream>>>(W, 2452112);
  prep_b1_kernel<<<(11 * 32 * 16 + 255) / 256, 256, 0, stream>>>(n1w2, n1b2, B1p);
  prep_b2_kernel<<<(6 * 2 * 32 * 16 + 255) / 256, 256, 0, stream>>>(n2w2, n2b2, B2p);
  edge1_kernel<<<2048, 32, 0, stream>>>(x, ei, ea, n1w1, n1b1, B1p, agg1, cnt);
  node1_kernel<<<(N_NODES * 16 + 255) / 256, 256, 0, stream>>>(x, agg1, cnt, root1, bias1, h1);
  edge2_kernel<<<2048, 32, 0, stream>>>(h1, ei, ea, n2w1, n2b1, B2p, agg2);
  node2_pool_kernel<<<(N_NODES * 32 + 255) / 256, 256, 0, stream>>>(h1, agg2, cnt, root2, bias2,
                                                                    batch, gsum, gcnt);
  head_kernel<<<1, 64, 0, stream>>>(gsum, gcnt, fc1w, fc1b, fc2w, fc2b, out);
}
